// Transformer_77979426226891
// MI455X (gfx1250) — matrix operand order repaired and checked by reading, not compiled, not run on hardware
//
#include <hip/hip_runtime.h>
#include <hip/hip_bf16.h>

// ---------------- problem constants (match reference) ----------------
constexpr int SEQ    = 2048;     // S_SRC == S_TGT
constexpr int DM     = 1024;     // D
constexpr int DFFN   = 4096;     // DFF
constexpr int NHEADS = 16;
constexpr int HDIM   = 64;
constexpr int VO     = 32000;    // V_OUT
constexpr int L_ENC  = 6;
constexpr int L_DEC  = 6;
constexpr float LN_EPS    = 1e-6f;
constexpr float MASK_MULT = -100000.0f;

typedef __attribute__((ext_vector_type(16))) _Float16 v16h;
typedef __attribute__((ext_vector_type(8)))  _Float16 v8h;
typedef __attribute__((ext_vector_type(8)))  float    v8f;

__device__ __forceinline__ v16h cat8(v8h lo, v8h hi) {
  return __builtin_shufflevector(lo, hi, 0, 1, 2, 3, 4, 5, 6, 7,
                                 8, 9, 10, 11, 12, 13, 14, 15);
}

// ---------------- WMMA fragment loaders from global (contiguous k = 16h + s order) -----
// A-matrix 16x32 f16: lanes 0-15 hold rows M (K 0-15),
// lanes 16-31 hold same rows (K 16-31).
__device__ __forceinline__ v16h load_a_frag(const float* __restrict__ A, int lda,
                                            int m, int k0, int lane) {
  const int khalf = (lane < 16) ? 0 : 16;
  const float4* p0 = (const float4*)(A + (size_t)m * lda + k0 + khalf);
  const float4* p1 = (const float4*)(A + (size_t)m * lda + k0 + 8 + khalf);
  float4 f0 = p0[0], f1 = p0[1], f2 = p1[0], f3 = p1[1];
  float t[16] = {f0.x, f0.y, f0.z, f0.w, f1.x, f1.y, f1.z, f1.w,
                 f2.x, f2.y, f2.z, f2.w, f3.x, f3.y, f3.z, f3.w};
  v16h a;
#pragma unroll
  for (int e = 0; e < 16; ++e) a[e] = (_Float16)t[e];
  return a;
}

// B-matrix 32x16 f16, B stored row-major [K, N]: lane = column n,
// lanes 0-15 carry K=0..15, lanes 16-31 carry K=16..31.
__device__ __forceinline__ v16h load_b_frag(const float* __restrict__ B, int ldb,
                                            int n, int k0, int lane) {
  const int kb = (lane < 16) ? 0 : 16;
  v16h b;
#pragma unroll
  for (int e = 0; e < 16; ++e)
    b[e] = (_Float16)B[(size_t)(k0 + kb + e) * ldb + n];
  return b;
}

// B(k,n) = Bt[n*ldb + k]  (i.e. Q @ K^T): contiguous along k.
__device__ __forceinline__ v16h load_bt_frag(const float* __restrict__ Bt, int ldb,
                                             int n, int k0, int lane) {
  const int kb = (lane < 16) ? 0 : 16;
  const float4* p = (const float4*)(Bt + (size_t)n * ldb + k0 + kb);
  float4 f0 = p[0], f1 = p[1], f2 = p[2], f3 = p[3];
  float t[16] = {f0.x, f0.y, f0.z, f0.w, f1.x, f1.y, f1.z, f1.w,
                 f2.x, f2.y, f2.z, f2.w, f3.x, f3.y, f3.z, f3.w};
  v16h b;
#pragma unroll
  for (int e = 0; e < 16; ++e) b[e] = (_Float16)t[e];
  return b;
}

// =====================================================================
// LDS-staged GEMM: C[M,N] = A[M,K] @ B[K,N] + bias (opt ReLU)
// 256 threads (8 waves). Macro tile 64(M) x 128(N); each wave computes a
// 32x32 quad = 4 WMMAs per 32-deep k-chunk, fragments served from LDS.
// =====================================================================
constexpr int LDS_STRIDE = 40;   // 32 halves + pad (bank spread), 80 B rows

template <int RELU>
__global__ void __launch_bounds__(256)
gemm_lds_kernel(const float* __restrict__ A, const float* __restrict__ B,
                const float* __restrict__ bias, float* __restrict__ C,
                int M, int N, int K) {
  __shared__ _Float16 As[64 * LDS_STRIDE];   // [m][k]   (5.1 KB)
  __shared__ _Float16 Bs[128 * LDS_STRIDE];  // [n][k]   (10.2 KB, transposed)

  const int tid  = threadIdx.x;
  const int lane = tid & 31;
  const int w    = tid >> 5;
  const int quadM = (w >> 2) * 32;            // 0 / 32
  const int quadN = (w & 3) * 32;             // 0 / 32 / 64 / 96
  const int blockM = blockIdx.y * 64;
  const int blockN = blockIdx.x * 128;

  const int khalf = (lane < 16) ? 0 : 16;     // A fragment k-phase
  const int kb    = (lane < 16) ? 0 : 16;     // B fragment k-phase
  const int l15   = lane & 15;

  // cooperative-load index maps
  const int ar  = tid >> 2;                   // A row 0..63
  const int ac4 = tid & 3;                    // A float4 column group

  v8f c00 = {}, c01 = {}, c10 = {}, c11 = {};

  for (int k0 = 0; k0 < K; k0 += 32) {
    // ---- stage A tile 64x32 (f32 -> f16) ----
#pragma unroll
    for (int i = 0; i < 2; ++i) {
      const int col = (ac4 + 4 * i) * 4;      // 0..28
      float4 f = *(const float4*)(A + (size_t)(blockM + ar) * K + k0 + col);
      _Float16* dst = &As[ar * LDS_STRIDE + col];
      dst[0] = (_Float16)f.x; dst[1] = (_Float16)f.y;
      dst[2] = (_Float16)f.z; dst[3] = (_Float16)f.w;
    }
    // ---- stage B tile 32x128 transposed (f32 -> f16) ----
#pragma unroll
    for (int i = 0; i < 4; ++i) {
      const int idx = tid + i * 256;          // 0..1023
      const int kk  = idx >> 5;               // 0..31
      const int c4  = (idx & 31) * 4;         // 0..124
      float4 f = *(const float4*)(B + (size_t)(k0 + kk) * N + blockN + c4);
      Bs[(c4 + 0) * LDS_STRIDE + kk] = (_Float16)f.x;
      Bs[(c4 + 1) * LDS_STRIDE + kk] = (_Float16)f.y;
      Bs[(c4 + 2) * LDS_STRIDE + kk] = (_Float16)f.z;
      Bs[(c4 + 3) * LDS_STRIDE + kk] = (_Float16)f.w;
    }
    // prefetch next k-chunk into cache while we compute
    if (k0 + 32 < K) {
      __builtin_prefetch(A + (size_t)(blockM + ar) * K + k0 + 32, 0, 0);
      __builtin_prefetch(B + (size_t)(k0 + 32 + (tid >> 3)) * N + blockN, 0, 0);
    }
    __syncthreads();

    // ---- fragments from LDS ----
    const int m0 = quadM + l15, m1 = m0 + 16;
    const int n0 = quadN + l15, n1 = n0 + 16;
    const v8h* pa0 = (const v8h*)&As[m0 * LDS_STRIDE + khalf];
    const v8h* qa0 = (const v8h*)&As[m0 * LDS_STRIDE + 8 + khalf];
    const v8h* pa1 = (const v8h*)&As[m1 * LDS_STRIDE + khalf];
    const v8h* qa1 = (const v8h*)&As[m1 * LDS_STRIDE + 8 + khalf];
    v16h a0 = cat8(pa0[0], qa0[0]);
    v16h a1 = cat8(pa1[0], qa1[0]);
    const v8h* pb0 = (const v8h*)&Bs[n0 * LDS_STRIDE + kb];
    const v8h* pb1 = (const v8h*)&Bs[n1 * LDS_STRIDE + kb];
    v16h b0 = cat8(pb0[0], pb0[1]);
    v16h b1 = cat8(pb1[0], pb1[1]);

    c00 = __builtin_amdgcn_wmma_f32_16x16x32_f16(false, a0, false, b0, (short)0, c00, false, false);
    c01 = __builtin_amdgcn_wmma_f32_16x16x32_f16(false, a0, false, b1, (short)0, c01, false, false);
    c10 = __builtin_amdgcn_wmma_f32_16x16x32_f16(false, a1, false, b0, (short)0, c10, false, false);
    c11 = __builtin_amdgcn_wmma_f32_16x16x32_f16(false, a1, false, b1, (short)0, c11, false, false);
    __syncthreads();
  }

  // ---- epilogue: 4 output tiles, D layout: VGPR j = rows j / j+8 ----
  const int mb = ((lane < 16) ? 0 : 8);
  const int nA = blockN + quadN + l15, nB = nA + 16;
  const float bvA = bias ? bias[nA] : 0.0f;
  const float bvB = bias ? bias[nB] : 0.0f;
#pragma unroll
  for (int j = 0; j < 8; ++j) {
    const int r0 = blockM + quadM + mb + j;
    const int r1 = r0 + 16;
    float v00 = c00[j] + bvA, v01 = c01[j] + bvB;
    float v10 = c10[j] + bvA, v11 = c11[j] + bvB;
    if (RELU) {
      v00 = fmaxf(v00, 0.0f); v01 = fmaxf(v01, 0.0f);
      v10 = fmaxf(v10, 0.0f); v11 = fmaxf(v11, 0.0f);
    }
    C[(size_t)r0 * N + nA] = v00;
    C[(size_t)r0 * N + nB] = v01;
    C[(size_t)r1 * N + nA] = v10;
    C[(size_t)r1 * N + nB] = v11;
  }
}

// ---------------- attention scores: S[h,q,k] = (Q_h @ K_h^T)/8 (+ causal) ----
__global__ void attn_scores_kernel(const float* __restrict__ Q,
                                   const float* __restrict__ Km,
                                   float* __restrict__ scores, int causal) {
  const int lane = threadIdx.x & 31;
  const int w    = threadIdx.x >> 5;
  const int h    = blockIdx.z;
  const int tileM = blockIdx.y * 32 + (w >> 2) * 16;
  const int tileN = blockIdx.x * 64 + (w & 3) * 16;
  const int m = tileM + (lane & 15);
  const int n = tileN + (lane & 15);
  const float* Qh = Q + h * HDIM;
  const float* Kh = Km + h * HDIM;

  v8f c = {};
#pragma unroll
  for (int k0 = 0; k0 < HDIM; k0 += 32) {
    v16h a = load_a_frag(Qh, DM, m, k0, lane);
    v16h b = load_bt_frag(Kh, DM, n, k0, lane);
    c = __builtin_amdgcn_wmma_f32_16x16x32_f16(false, a, false, b,
                                               (short)0, c, false, false);
  }
  float* out = scores + (size_t)h * SEQ * SEQ;
  const int mbase = tileM + ((lane < 16) ? 0 : 8);
#pragma unroll
  for (int j = 0; j < 8; ++j) {
    const int row = mbase + j;
    float v = c[j] * 0.125f;                 // 1/sqrt(HDIM)
    if (causal && n > row) v += MASK_MULT;   // (1 - tril) * MASK_MULT
    out[(size_t)row * SEQ + n] = v;
  }
}

// ---------------- attention context: O[:, h*64 + n] = P_h @ V_h ----------------
__global__ void attn_ctx_kernel(const float* __restrict__ P,
                                const float* __restrict__ V,
                                float* __restrict__ O) {
  const int lane = threadIdx.x & 31;
  const int w    = threadIdx.x >> 5;
  const int h    = blockIdx.z;
  const int tileM = blockIdx.y * 32 + (w >> 2) * 16;
  const int tileN = (w & 3) * 16;          // N = HDIM = 64 = 4 wave tiles
  const int m = tileM + (lane & 15);
  const int n = tileN + (lane & 15);
  const float* A  = P + (size_t)h * SEQ * SEQ;
  const float* Vh = V + h * HDIM;

  v8f c = {};
  for (int k0 = 0; k0 < SEQ; k0 += 32) {
    v16h a = load_a_frag(A, SEQ, m, k0, lane);
    v16h b = load_b_frag(Vh, DM, n, k0, lane);
    c = __builtin_amdgcn_wmma_f32_16x16x32_f16(false, a, false, b,
                                               (short)0, c, false, false);
  }
  const int mbase = tileM + ((lane < 16) ? 0 : 8);
#pragma unroll
  for (int j = 0; j < 8; ++j)
    O[(size_t)(mbase + j) * DM + h * HDIM + n] = c[j];
}

// ---------------- embedding + positional encoding ----------------
__global__ void embed_pos_kernel(const int* __restrict__ ids,
                                 const float* __restrict__ emb,
                                 float* __restrict__ X) {
  const int s  = blockIdx.x;
  const int id = ids[s];
  const float pos = (float)(s + 1);
  for (int d = threadIdx.x; d < DM; d += blockDim.x) {
    float angle = __powf(10000.0f, 2.0f * (float)(d / 2) / (float)DM);
    float arg   = pos / angle;
    float pe    = (d & 1) ? __cosf(arg) : __sinf(arg);
    X[(size_t)s * DM + d] = emb[(size_t)id * DM + d] + pe;
  }
}

// ---------------- LayerNorm(a + b) * g + beta, one row per block --------------
__global__ void ln_residual_kernel(const float* __restrict__ a,
                                   const float* __restrict__ b,
                                   const float* __restrict__ g,
                                   const float* __restrict__ beta,
                                   float* __restrict__ out) {
  __shared__ float red[256];
  const int row = blockIdx.x;
  const int tid = threadIdx.x;
  const float* pa = a + (size_t)row * DM;
  const float* pb = b + (size_t)row * DM;

  float vals[DM / 256];
  float s = 0.0f;
#pragma unroll
  for (int i = 0; i < DM / 256; ++i) {
    const int d = tid + i * 256;
    vals[i] = pa[d] + pb[d];
    s += vals[i];
  }
  red[tid] = s; __syncthreads();
  for (int off = 128; off > 0; off >>= 1) {
    if (tid < off) red[tid] += red[tid + off];
    __syncthreads();
  }
  const float mean = red[0] / (float)DM;
  __syncthreads();

  float vs = 0.0f;
#pragma unroll
  for (int i = 0; i < DM / 256; ++i) {
    const float d = vals[i] - mean;
    vs += d * d;
  }
  red[tid] = vs; __syncthreads();
  for (int off = 128; off > 0; off >>= 1) {
    if (tid < off) red[tid] += red[tid + off];
    __syncthreads();
  }
  const float inv = rsqrtf(red[0] / (float)DM + LN_EPS);

#pragma unroll
  for (int i = 0; i < DM / 256; ++i) {
    const int d = tid + i * 256;
    out[(size_t)row * DM + d] = g[d] * (vals[i] - mean) * inv + beta[d];
  }
}

// ---------------- row-wise softmax (in place), one row per block --------------
__global__ void softmax_rows_kernel(float* __restrict__ x, int cols) {
  __shared__ float red[256];
  const int tid = threadIdx.x;
  const size_t base = (size_t)blockIdx.x * cols;

  float mx = -3.4e38f;
  for (int c = tid; c < cols; c += 256) mx = fmaxf(mx, x[base + c]);
  red[tid] = mx; __syncthreads();
  for (int off = 128; off > 0; off >>= 1) {
    if (tid < off) red[tid] = fmaxf(red[tid], red[tid + off]);
    __syncthreads();
  }
  const float rowmax = red[0];
  __syncthreads();

  float sum = 0.0f;
  for (int c = tid; c < cols; c += 256) {
    float e = __expf(x[base + c] - rowmax);
    x[base + c] = e;
    sum += e;
  }
  red[tid] = sum; __syncthreads();
  for (int off = 128; off > 0; off >>= 1) {
    if (tid < off) red[tid] += red[tid + off];
    __syncthreads();
  }
  const float inv = 1.0f / red[0];
  for (int c = tid; c < cols; c += 256) x[base + c] *= inv;
}

// =====================================================================
//                          host orchestration
// =====================================================================
extern "C" void kernel_launch(void* const* d_in, const int* in_sizes, int n_in,
                              void* d_out, int out_size, void* d_ws, size_t ws_size,
                              hipStream_t stream) {
  (void)in_sizes; (void)n_in; (void)out_size; (void)ws_size;

  // ---- inputs (setup_inputs() insertion order, nested dicts flattened) ----
  const int*   src_ids = (const int*)d_in[0];
  const int*   tgt_ids = (const int*)d_in[1];
  const float* enc_emb = (const float*)d_in[2];
  const float* dec_emb = (const float*)d_in[3];

  const float* e_wq_W = (const float*)d_in[4];  const float* e_wq_b = (const float*)d_in[5];
  const float* e_wk_W = (const float*)d_in[6];  const float* e_wk_b = (const float*)d_in[7];
  const float* e_wv_W = (const float*)d_in[8];  const float* e_wv_b = (const float*)d_in[9];
  const float* e_wo_W = (const float*)d_in[10]; const float* e_wo_b = (const float*)d_in[11];
  const float* e_W1   = (const float*)d_in[12]; const float* e_b1   = (const float*)d_in[13];
  const float* e_W2   = (const float*)d_in[14]; const float* e_b2   = (const float*)d_in[15];
  const float* e_ln1g = (const float*)d_in[16]; const float* e_ln1b = (const float*)d_in[17];
  const float* e_ln2g = (const float*)d_in[18]; const float* e_ln2b = (const float*)d_in[19];

  const float* dsa_wq_W = (const float*)d_in[20]; const float* dsa_wq_b = (const float*)d_in[21];
  const float* dsa_wk_W = (const float*)d_in[22]; const float* dsa_wk_b = (const float*)d_in[23];
  const float* dsa_wv_W = (const float*)d_in[24]; const float* dsa_wv_b = (const float*)d_in[25];
  const float* dsa_wo_W = (const float*)d_in[26]; const float* dsa_wo_b = (const float*)d_in[27];
  const float* dca_wq_W = (const float*)d_in[28]; const float* dca_wq_b = (const float*)d_in[29];
  const float* dca_wk_W = (const float*)d_in[30]; const float* dca_wk_b = (const float*)d_in[31];
  const float* dca_wv_W = (const float*)d_in[32]; const float* dca_wv_b = (const float*)d_in[33];
  const float* dca_wo_W = (const float*)d_in[34]; const float* dca_wo_b = (const float*)d_in[35];
  const float* d_W1   = (const float*)d_in[36]; const float* d_b1   = (const float*)d_in[37];
  const float* d_W2   = (const float*)d_in[38]; const float* d_b2   = (const float*)d_in[39];
  const float* d_ln1g = (const float*)d_in[40]; const float* d_ln1b = (const float*)d_in[41];
  const float* d_ln2g = (const float*)d_in[42]; const float* d_ln2b = (const float*)d_in[43];
  const float* d_ln3g = (const float*)d_in[44]; const float* d_ln3b = (const float*)d_in[45];

  const float* final_W = (const float*)d_in[46];
  const float* final_b = (const float*)d_in[47];

  // ---- workspace carve ----
  const size_t SD = (size_t)SEQ * DM;
  float* ws = (float*)d_ws;
  float* X    = ws; ws += SD;                  // encoder activations / enc_out
  float* Y    = ws; ws += SD;                  // decoder activations
  float* N1   = ws; ws += SD;
  float* N2   = ws; ws += SD;
  float* T1   = ws; ws += SD;                  // attn context / ffn out
  float* T2   = ws; ws += SD;                  // attn output projection
  float* Qb   = ws; ws += SD;
  float* Kb   = ws; ws += SD;
  float* Vb   = ws; ws += SD;
  float* FH   = ws; ws += (size_t)SEQ * DFFN;  // ffn hidden
  float* SC   = ws; ws += (size_t)NHEADS * SEQ * SEQ;  // attention scores/probs

  // ---- outputs: (probs, mw0, cw_last) concatenated flat ----
  float* out_probs = (float*)d_out;
  float* out_mw0   = out_probs + (size_t)SEQ * VO;
  float* out_cw    = out_mw0 + (size_t)NHEADS * SEQ * SEQ;

  // ---- helpers ----
  auto gemm = [&](const float* A, const float* B, const float* bias, float* C,
                  int M, int N, int K, bool relu) {
    dim3 grid(N / 128, M / 64);
    if (relu)
      gemm_lds_kernel<1><<<grid, 256, 0, stream>>>(A, B, bias, C, M, N, K);
    else
      gemm_lds_kernel<0><<<grid, 256, 0, stream>>>(A, B, bias, C, M, N, K);
  };

  auto attention = [&](const float* qx, const float* kvx,
                       const float* wqW, const float* wqb,
                       const float* wkW, const float* wkb,
                       const float* wvW, const float* wvb,
                       const float* woW, const float* wob,
                       int layer, bool causal, float* outbuf, float* save_w) {
    const size_t WO = (size_t)layer * DM * DM;
    const size_t BO = (size_t)layer * DM;
    gemm(qx,  wqW + WO, wqb + BO, Qb, SEQ, DM, DM, false);
    gemm(kvx, wkW + WO, wkb + BO, Kb, SEQ, DM, DM, false);
    gemm(kvx, wvW + WO, wvb + BO, Vb, SEQ, DM, DM, false);

    attn_scores_kernel<<<dim3(SEQ / 64, SEQ / 32, NHEADS), 256, 0, stream>>>(
        Qb, Kb, SC, causal ? 1 : 0);
    softmax_rows_kernel<<<NHEADS * SEQ, 256, 0, stream>>>(SC, SEQ);
    if (save_w)
      hipMemcpyAsync(save_w, SC, (size_t)NHEADS * SEQ * SEQ * sizeof(float),
                     hipMemcpyDeviceToDevice, stream);
    attn_ctx_kernel<<<dim3(1, SEQ / 32, NHEADS), 256, 0, stream>>>(SC, Vb, T1);
    gemm(T1, woW + WO, wob + BO, outbuf, SEQ, DM, DM, false);
  };

  // =================== encoder ===================
  embed_pos_kernel<<<SEQ, 256, 0, stream>>>(src_ids, enc_emb, X);
  for (int i = 0; i < L_ENC; ++i) {
    attention(X, X, e_wq_W, e_wq_b, e_wk_W, e_wk_b, e_wv_W, e_wv_b,
              e_wo_W, e_wo_b, i, false, T2, nullptr);
    ln_residual_kernel<<<SEQ, 256, 0, stream>>>(
        X, T2, e_ln1g + (size_t)i * DM, e_ln1b + (size_t)i * DM, X);
    gemm(X, e_W1 + (size_t)i * DM * DFFN, e_b1 + (size_t)i * DFFN, FH,
         SEQ, DFFN, DM, true);
    gemm(FH, e_W2 + (size_t)i * DFFN * DM, e_b2 + (size_t)i * DM, T1,
         SEQ, DM, DFFN, false);
    ln_residual_kernel<<<SEQ, 256, 0, stream>>>(
        T1, X, e_ln2g + (size_t)i * DM, e_ln2b + (size_t)i * DM, X);
  }
  // enc_out lives in X for the rest of the pass.

  // =================== decoder ===================
  embed_pos_kernel<<<SEQ, 256, 0, stream>>>(tgt_ids, dec_emb, Y);
  for (int i = 0; i < L_DEC; ++i) {
    attention(Y, Y, dsa_wq_W, dsa_wq_b, dsa_wk_W, dsa_wk_b, dsa_wv_W, dsa_wv_b,
              dsa_wo_W, dsa_wo_b, i, /*causal=*/true, T2,
              (i == 0) ? out_mw0 : nullptr);
    ln_residual_kernel<<<SEQ, 256, 0, stream>>>(
        T2, Y, d_ln1g + (size_t)i * DM, d_ln1b + (size_t)i * DM, N1);

    attention(N1, X, dca_wq_W, dca_wq_b, dca_wk_W, dca_wk_b, dca_wv_W, dca_wv_b,
              dca_wo_W, dca_wo_b, i, /*causal=*/false, T2,
              (i == L_DEC - 1) ? out_cw : nullptr);
    ln_residual_kernel<<<SEQ, 256, 0, stream>>>(
        T2, N1, d_ln2g + (size_t)i * DM, d_ln2b + (size_t)i * DM, N2);

    gemm(N2, d_W1 + (size_t)i * DM * DFFN, d_b1 + (size_t)i * DFFN, FH,
         SEQ, DFFN, DM, true);
    gemm(FH, d_W2 + (size_t)i * DFFN * DM, d_b2 + (size_t)i * DM, T1,
         SEQ, DM, DFFN, false);
    ln_residual_kernel<<<SEQ, 256, 0, stream>>>(
        T1, N2, d_ln3g + (size_t)i * DM, d_ln3b + (size_t)i * DM, Y);
  }

  // =================== final projection + softmax ===================
  gemm(Y, final_W, final_b, out_probs, SEQ, VO, DM, false);
  softmax_rows_kernel<<<SEQ, 256, 0, stream>>>(out_probs, VO);
}
